// Bottleneck_75325136437779
// MI455X (gfx1250) — matrix operand order repaired and checked by reading, not compiled, not run on hardware
//
#include <hip/hip_runtime.h>

#define DEVINL __device__ __forceinline__

typedef __attribute__((ext_vector_type(16))) __bf16 v16bf;
typedef __attribute__((ext_vector_type(8)))  float  v8f;
typedef __attribute__((ext_vector_type(4)))  float  f32x4;
typedef __attribute__((ext_vector_type(8)))  unsigned short us8;
typedef __attribute__((ext_vector_type(16))) unsigned short us16;
typedef __attribute__((ext_vector_type(4)))  unsigned int  u32x4;
typedef __attribute__((ext_vector_type(4)))  int           i32x4;
typedef __attribute__((ext_vector_type(8)))  int           i32x8;

// ---------- bf16 helpers (bit-exact, no __bf16 arithmetic) ----------
DEVINL unsigned short f2bf(float f){
  unsigned int u = __builtin_bit_cast(unsigned int, f);
  u = (u + 0x7FFFu + ((u >> 16) & 1u)) >> 16;      // round-to-nearest-even
  return (unsigned short)u;
}
DEVINL float bf2f(unsigned short s){
  unsigned int u = ((unsigned int)s) << 16;
  return __builtin_bit_cast(float, u);
}
DEVINL __bf16 us2bf(unsigned short s){ return __builtin_bit_cast(__bf16, s); }

DEVINL v8f wmma_bf16(v16bf a, v16bf b, v8f c){
  return __builtin_amdgcn_wmma_f32_16x16x32_bf16(false, a, false, b, (short)0, c, false, false);
}

// ---------- vectorized fragment loaders ----------
// A fragment: 16(M) x 32(K) bf16 row-major, row stride `stride` (elements).
DEVINL v16bf frag_a(const unsigned short* base, int stride, int lane){
  const unsigned short* p = base + (lane & 15) * stride + (lane >> 4) * 16;
  us8 lo = *reinterpret_cast<const us8*>(p);
  us8 hi = *reinterpret_cast<const us8*>(p + 8);
  us16 w = __builtin_shufflevector(lo, hi, 0,1,2,3,4,5,6,7,8,9,10,11,12,13,14,15);
  return __builtin_bit_cast(v16bf, w);
}
// B fragment from K-CONTIGUOUS layout: element (k,n) at base + n*nstride + k.
DEVINL v16bf frag_bt(const unsigned short* base, int nstride, int lane){
  const unsigned short* p = base + (lane & 15) * nstride + (lane >> 4) * 16;
  us8 lo = *reinterpret_cast<const us8*>(p);
  us8 hi = *reinterpret_cast<const us8*>(p + 8);
  us16 w = __builtin_shufflevector(lo, hi, 0,1,2,3,4,5,6,7,8,9,10,11,12,13,14,15);
  return __builtin_bit_cast(v16bf, w);
}
// B fragment, generic strided
DEVINL v16bf frag_b(const unsigned short* base, int kstride, int nstride, int lane){
  int n = lane & 15, k0 = (lane >> 4) * 16;
  const unsigned short* p = base + k0 * kstride + n * nstride;
  v16bf b;
#pragma unroll
  for (int e = 0; e < 16; ++e) b[e] = us2bf(p[e * kstride]);
  return b;
}
// B fragment from fp32 source, K contiguous (converted on the fly)
DEVINL v16bf frag_b_f32c(const float* base, int nstride, int lane){
  const float* p = base + (lane & 15) * nstride + (lane >> 4) * 16;
  v16bf b;
#pragma unroll
  for (int e = 0; e < 16; ++e) b[e] = us2bf(f2bf(p[e]));
  return b;
}

// ---------- TDM (Tensor Data Mover) support ----------
#if defined(__gfx1250__) && __has_builtin(__builtin_amdgcn_tensor_load_to_lds)
#define HAVE_TDM 1
DEVINL u32x4 tdm_g0(unsigned lds_addr, const void* gptr){
  unsigned long long ga = (unsigned long long)gptr;
  u32x4 g;
  g[0] = 1u;                                               // count=1, user mode
  g[1] = lds_addr;                                         // bits 63:32
  g[2] = (unsigned)(ga & 0xFFFFFFFFu);                     // global_addr[31:0]
  g[3] = (unsigned)((ga >> 32) & 0x1FFFFFFu) | (2u << 30); // addr[56:32] | type=2
  return g;
}
DEVINL i32x8 tdm_g1(int tile0, int tile1, unsigned dim0, unsigned dim1,
                    unsigned long long stride0, int dsize_code){
  i32x8 g;
  g[0] = (dsize_code & 3) << 16;
  g[1] = (int)((dim0 & 0xFFFFu) << 16);
  g[2] = (int)(((dim0 >> 16) & 0xFFFFu) | ((dim1 & 0xFFFFu) << 16));
  g[3] = (int)(((dim1 >> 16) & 0xFFFFu) | ((tile0 & 0xFFFF) << 16));
  g[4] = (int)(tile1 & 0xFFFF);
  g[5] = (int)(stride0 & 0xFFFFFFFFull);
  g[6] = (int)((stride0 >> 32) & 0xFFFFull);
  g[7] = 0;
  return g;
}
DEVINL void tdm_load(u32x4 g0, i32x8 g1){
#if __clang_major__ >= 23
  __builtin_amdgcn_tensor_load_to_lds(g0, g1, (i32x4)0, (i32x4)0, (i32x8)0, 0);
#else
  __builtin_amdgcn_tensor_load_to_lds(g0, g1, (i32x4)0, (i32x4)0, 0);
#endif
}
#endif

// ---------- problem constants ----------
#define NB     32
#define CIN    512
#define CMID   128
#define NPOS   4096
#define WDIM   64
#define DIMK   16
#define DIMV   32
#define CQ     64
#define COUT   512

// ---------- workspace layout (bytes) ----------
#define OFF_W1   ((size_t)0)
#define OFF_WQKV ((size_t)131072)
#define OFF_W3   ((size_t)159744)
#define OFF_EMB  ((size_t)290816)
#define OFF_Q    ((size_t)299008)
#define OFF_K    ((size_t)17076224)
#define OFF_VF   ((size_t)21270528)
#define OFF_CL   ((size_t)38047744)
#define OFF_Y    ((size_t)38113280)

// =====================================================================
// Kernel 0: weights -> bf16 (W1, Wqkv concat, W3, padded EMB)
// =====================================================================
__global__ __launch_bounds__(256) void k_prep(
    const float* __restrict__ w1, const float* __restrict__ qw,
    const float* __restrict__ kw, const float* __restrict__ vw,
    const float* __restrict__ w3, const float* __restrict__ emb,
    unsigned short* __restrict__ w1bf, unsigned short* __restrict__ wqkvbf,
    unsigned short* __restrict__ w3bf, unsigned short* __restrict__ embbf)
{
  int t = blockIdx.x * 256 + threadIdx.x;
  if (t < 65536){
    w1bf[t] = f2bf(w1[t]);
  } else if (t < 65536 + 14336){
    int i = t - 65536, row = i >> 7, c = i & 127;
    float f;
    if (row < 64)      f = qw[row * 128 + c];
    else if (row < 80) f = kw[(row - 64) * 128 + c];
    else               f = vw[(row - 80) * 128 + c];
    wqkvbf[i] = f2bf(f);
  } else if (t < 65536 + 14336 + 65536){
    int i = t - 65536 - 14336;
    w3bf[i] = f2bf(w3[i]);
  } else if (t < 65536 + 14336 + 65536 + 4096){
    int i = t - 65536 - 14336 - 65536;
    int k = i >> 8, tap = i & 255;
    embbf[i] = (tap < 225) ? f2bf(emb[k * 225 + tap]) : (unsigned short)0;
  }
}

// =====================================================================
// Kernel 1: conv1+bn1+relu -> (LDS h) -> qkv convs + bn(q), bn(v), softmax(k)
// =====================================================================
__global__ __launch_bounds__(256) void k_fused_qkv(
    const float* __restrict__ x,
    const unsigned short* __restrict__ w1bf,
    const unsigned short* __restrict__ wqkvbf,
    const float* __restrict__ g1, const float* __restrict__ b1,
    const float* __restrict__ m1, const float* __restrict__ v1,
    const float* __restrict__ gq, const float* __restrict__ bq,
    const float* __restrict__ mq, const float* __restrict__ vq,
    const float* __restrict__ gv, const float* __restrict__ bv,
    const float* __restrict__ mv, const float* __restrict__ vv,
    unsigned short* __restrict__ qws, unsigned short* __restrict__ kws,
    float* __restrict__ vfws)
{
  __shared__ __attribute__((aligned(32))) unsigned short sX[32 * 32];   // [n][k]
  __shared__ __attribute__((aligned(32))) unsigned short sHT[32 * 128]; // [n][c]
  __shared__ float sK[16 * 32];
  __shared__ float sS1[128], sB1[128], sSq[64], sBq[64], sSv[32], sBv[32];

  const int t = threadIdx.x, lane = t & 31, wave = t >> 5;
  const int b = blockIdx.x >> 7, n0 = (blockIdx.x & 127) * 32;

  if (t < 128){ float s = g1[t] * rsqrtf(v1[t] + 1e-5f); sS1[t] = s; sB1[t] = b1[t] - m1[t] * s; }
  else if (t < 192){ int c = t - 128; float s = gq[c] * rsqrtf(vq[c] + 1e-5f); sSq[c] = s; sBq[c] = bq[c] - mq[c] * s; }
  else if (t < 224){ int c = t - 192; float s = gv[c] * rsqrtf(vv[c] + 1e-5f); sSv[c] = s; sBv[c] = bv[c] - mv[c] * s; }

  // per-thread tile-load mapping: one b128 (4 floats) per thread per chunk
  const int lkk = t >> 3;              // row (channel within chunk) 0..31
  const int lnn = (t & 7) * 4;         // 4 consecutive positions

  // ---- GEMM1: h(128x32) = W1(128x512) @ X(512x32) ----
  v8f acc0 = (v8f)0.0f, acc1 = (v8f)0.0f;
  for (int kc = 0; kc < 16; ++kc){
    __syncthreads();
    {
      size_t gidx = ((size_t)(b * CIN + kc * 32 + lkk)) * NPOS + n0 + lnn;
      f32x4 xv = *reinterpret_cast<const f32x4*>(&x[gidx]);
#pragma unroll
      for (int j = 0; j < 4; ++j)
        sX[(lnn + j) * 32 + lkk] = f2bf(xv[j]);        // transposed: K contiguous
      if (kc + 1 < 16)
        __builtin_prefetch(&x[gidx + (size_t)32 * NPOS], 0, 3);  // near-cache
    }
    __syncthreads();
    v16bf a   = frag_a(w1bf + (wave * 16) * 512 + kc * 32, 512, lane);
    v16bf b0  = frag_bt(sX, 32, lane);
    v16bf b1v = frag_bt(sX + 16 * 32, 32, lane);
    acc0 = wmma_bf16(a, b0, acc0);
    acc1 = wmma_bf16(a, b1v, acc1);
  }
  __syncthreads();
  {
    int half = lane >> 4;
#pragma unroll
    for (int nt = 0; nt < 2; ++nt){
      int col = nt * 16 + (lane & 15);
#pragma unroll
      for (int r = 0; r < 8; ++r){
        int m = wave * 16 + r + 8 * half;
        float f = (nt ? acc1[r] : acc0[r]) * sS1[m] + sB1[m];
        f = f > 0.0f ? f : 0.0f;
        sHT[col * 128 + m] = f2bf(f);                  // transposed: C contiguous
      }
    }
  }
  __syncthreads();

  // ---- GEMM2: t(112x32) = Wqkv(112x128) @ h(128x32) ----
  if (wave < 7){
    v8f q0 = (v8f)0.0f, q1 = (v8f)0.0f;
#pragma unroll
    for (int kc = 0; kc < 4; ++kc){
      v16bf a   = frag_a(wqkvbf + (wave * 16) * 128 + kc * 32, 128, lane);
      v16bf b0  = frag_bt(sHT + kc * 32, 128, lane);
      v16bf b1v = frag_bt(sHT + 16 * 128 + kc * 32, 128, lane);
      q0 = wmma_bf16(a, b0, q0);
      q1 = wmma_bf16(a, b1v, q1);
    }
    int half = lane >> 4;
#pragma unroll
    for (int nt = 0; nt < 2; ++nt){
      int nn = nt * 16 + (lane & 15), n = n0 + nn;
#pragma unroll
      for (int r = 0; r < 8; ++r){
        int gm = wave * 16 + r + 8 * half;
        float f = nt ? q1[r] : q0[r];
        if (gm < 64){
          f = f * sSq[gm] + sBq[gm];
          qws[((size_t)(b * CQ + gm)) * NPOS + n] = f2bf(f);
        } else if (gm < 80){
          sK[(gm - 64) * 32 + nn] = f;
        } else {
          int c = gm - 80;
          f = f * sSv[c] + sBv[c];
          vfws[((size_t)(b * DIMV + c)) * NPOS + n] = f;
        }
      }
    }
  }
  __syncthreads();
  if (t < 32){
    int n = n0 + t;
    float mx = -1e30f;
#pragma unroll
    for (int j = 0; j < 16; ++j) mx = fmaxf(mx, sK[j * 32 + t]);
    float e[16], s = 0.0f;
#pragma unroll
    for (int j = 0; j < 16; ++j){ e[j] = __expf(sK[j * 32 + t] - mx); s += e[j]; }
    float inv = 1.0f / s;
#pragma unroll
    for (int j = 0; j < 16; ++j)
      kws[((size_t)(b * DIMK + j)) * NPOS + n] = f2bf(e[j] * inv);
  }
}

// =====================================================================
// Kernel 2: content_lambda[b] (16x32) = ksoft(16xN) @ v^T(Nx32)
// =====================================================================
__global__ __launch_bounds__(256) void k_content_lambda(
    const unsigned short* __restrict__ kws, const float* __restrict__ vfws,
    float* __restrict__ clws)
{
  __shared__ float sCL[16 * 32];
  const int t = threadIdx.x, lane = t & 31, wave = t >> 5, b = blockIdx.x;
  sCL[t] = 0.0f; sCL[t + 256] = 0.0f;
  __syncthreads();

  v8f a0 = (v8f)0.0f, a1 = (v8f)0.0f;
  const unsigned short* kbase = kws + (size_t)b * DIMK * NPOS;
  const float* vbase = vfws + (size_t)b * DIMV * NPOS;
  for (int ch = wave; ch < 128; ch += 8){
    int nc = ch * 32;
    v16bf a  = frag_a(kbase + nc, NPOS, lane);
    v16bf b0 = frag_b_f32c(vbase + nc, NPOS, lane);
    v16bf b1 = frag_b_f32c(vbase + (size_t)16 * NPOS + nc, NPOS, lane);
    a0 = wmma_bf16(a, b0, a0);
    a1 = wmma_bf16(a, b1, a1);
  }
  int half = lane >> 4;
#pragma unroll
  for (int nt = 0; nt < 2; ++nt){
    int v = nt * 16 + (lane & 15);
#pragma unroll
    for (int r = 0; r < 8; ++r){
      int k = r + 8 * half;
      atomicAdd(&sCL[k * 32 + v], nt ? a1[r] : a0[r]);
    }
  }
  __syncthreads();
  clws[b * 512 + t]       = sCL[t];
  clws[b * 512 + 256 + t] = sCL[256 + t];
}

// =====================================================================
// Kernel 3: pos conv (implicit GEMM via WMMA) + lambda apply + bn2 + relu
// EMB A-fragments hoisted into registers once per block (full A reuse).
// =====================================================================
__global__ __launch_bounds__(256) void k_lambda_apply(
    const unsigned short* __restrict__ qws, const float* __restrict__ vfws,
    const unsigned short* __restrict__ embbf, const float* __restrict__ clws,
    const float* __restrict__ g2, const float* __restrict__ b2,
    const float* __restrict__ m2, const float* __restrict__ v2,
    unsigned short* __restrict__ yws)
{
  __shared__ __attribute__((aligned(32))) unsigned short sV[8 * 30 * 32];
  __shared__ __attribute__((aligned(32))) unsigned short sQ[64 * 256];
  __shared__ __attribute__((aligned(32))) unsigned short sEmb[16 * 256];
  __shared__ float sCL[512];
  __shared__ float sS2[128], sB2[128];

  const int t = threadIdx.x, lane = t & 31, wave = t >> 5;
  const int b = blockIdx.x >> 4, tile = blockIdx.x & 15;
  const int py0 = (tile >> 2) * 16, px0 = (tile & 3) * 16;

  for (int i = t; i < 4096; i += 256) sEmb[i] = embbf[i];
  for (int i = t; i < 16384; i += 256){
    int c = i >> 8, p = i & 255, py = p >> 4, px = p & 15;
    sQ[i] = qws[((size_t)(b * CQ + c)) * NPOS + (py0 + py) * WDIM + px0 + px];
  }
  for (int i = t; i < 512; i += 256) sCL[i] = clws[b * 512 + i];
  if (t < 128){ float s = g2[t] * rsqrtf(v2[t] + 1e-5f); sS2[t] = s; sB2[t] = b2[t] - m2[t] * s; }
  __syncthreads();

  // hoist the 8 EMB A-fragments into registers (invariant for whole block)
  v16bf aF[8];
#pragma unroll
  for (int kc = 0; kc < 8; ++kc) aF[kc] = frag_a(sEmb + kc * 32, 256, lane);

  const int col = lane & 15, half = lane >> 4;

  for (int vg = 0; vg < 4; ++vg){
    __syncthreads();
    for (int i = t; i < 7200; i += 256){
      int c = i / 900, rem = i % 900, yy = rem / 30, xx = rem % 30;
      int gy = py0 - 7 + yy, gx = px0 - 7 + xx;
      float f = 0.0f;
      if (gy >= 0 && gy < WDIM && gx >= 0 && gx < WDIM)
        f = vfws[((size_t)(b * DIMV + vg * 8 + c)) * NPOS + gy * WDIM + gx];
      sV[c * 960 + yy * 32 + xx] = f2bf(f);
    }
    __syncthreads();

    for (int vl = 0; vl < 8; ++vl){
      const unsigned short* vp = sV + vl * 960;
      int vglob = vg * 8 + vl;
#pragma unroll
      for (int u = 0; u < 2; ++u){
        int nt = wave * 2 + u;
        int p = nt * 16 + col, py = p >> 4, px = p & 15;
        v8f acc = (v8f)0.0f;
#pragma unroll
        for (int kc = 0; kc < 8; ++kc){
          v16bf bb;
          int kbase = kc * 32 + half * 16;
#pragma unroll
          for (int e = 0; e < 16; ++e){
            int tap = kbase + e;
            unsigned short uval = 0;
            if (tap < 225){
              int dy = tap / 15, dx = tap - dy * 15;
              uval = vp[(py + dy) * 32 + (px + dx)];
            }
            bb[e] = us2bf(uval);
          }
          acc = wmma_bf16(aF[kc], bb, acc);
        }
        float s0 = 0.f, s1 = 0.f, s2 = 0.f, s3 = 0.f;
#pragma unroll
        for (int r = 0; r < 8; ++r){
          int k = r + 8 * half;
          float lam = sCL[k * 32 + vglob] + acc[r];
          s0 += bf2f(sQ[(0 * 16 + k) * 256 + p]) * lam;
          s1 += bf2f(sQ[(1 * 16 + k) * 256 + p]) * lam;
          s2 += bf2f(sQ[(2 * 16 + k) * 256 + p]) * lam;
          s3 += bf2f(sQ[(3 * 16 + k) * 256 + p]) * lam;
        }
        s0 += __shfl_xor(s0, 16, 32); s1 += __shfl_xor(s1, 16, 32);
        s2 += __shfl_xor(s2, 16, 32); s3 += __shfl_xor(s3, 16, 32);
        if (half == 0){
          int n = (py0 + py) * WDIM + px0 + px;
          float ss[4] = {s0, s1, s2, s3};
#pragma unroll
          for (int hd = 0; hd < 4; ++hd){
            int c = hd * 32 + vglob;
            float f = ss[hd] * sS2[c] + sB2[c];
            f = f > 0.0f ? f : 0.0f;
            yws[((size_t)(b * CMID + c)) * NPOS + n] = f2bf(f);
          }
        }
      }
    }
  }
}

// =====================================================================
// Kernel 4: conv3 + bn3 + residual + relu. Y tiles staged via TDM when
// available (tensor_load_to_lds + s_wait_tensorcnt), else cooperative load.
// =====================================================================
__global__ __launch_bounds__(256) void k_conv3(
    const unsigned short* __restrict__ yws, const unsigned short* __restrict__ w3bf,
    const float* __restrict__ xin,
    const float* __restrict__ g3, const float* __restrict__ b3,
    const float* __restrict__ m3, const float* __restrict__ v3,
    float* __restrict__ out)
{
  __shared__ __attribute__((aligned(32))) unsigned short sY[32 * 32]; // [k][n]
  __shared__ float sS3[512], sB3[512];
  const int t = threadIdx.x, lane = t & 31, wave = t >> 5;
  const int b = blockIdx.x >> 7, n0 = (blockIdx.x & 127) * 32;

  {
    int c = t;       float s = g3[c] * rsqrtf(v3[c] + 1e-5f); sS3[c] = s; sB3[c] = b3[c] - m3[c] * s;
    c = t + 256;           s = g3[c] * rsqrtf(v3[c] + 1e-5f); sS3[c] = s; sB3[c] = b3[c] - m3[c] * s;
  }

  v8f acc[8];
#pragma unroll
  for (int i = 0; i < 8; ++i) acc[i] = (v8f)0.0f;

  for (int kc = 0; kc < 4; ++kc){
    __syncthreads();
#if defined(HAVE_TDM)
    if (t < 32){
      const unsigned short* gsrc = yws + ((size_t)(b * CMID + kc * 32)) * NPOS + n0;
      unsigned lds_addr = (unsigned)(unsigned long long)(void*)&sY[0];
      tdm_load(tdm_g0(lds_addr, (const void*)gsrc),
               tdm_g1(/*tile0=*/32, /*tile1=*/32, /*dim0=*/NPOS, /*dim1=*/CMID,
                      /*stride0=*/NPOS, /*dsize=*/1));
      __builtin_amdgcn_s_wait_tensorcnt(0);
    }
#else
#pragma unroll
    for (int i = 0; i < 4; ++i){
      int e = t + i * 256, kk = e >> 5, nn = e & 31;
      sY[e] = yws[((size_t)(b * CMID + kc * 32 + kk)) * NPOS + n0 + nn];
    }
#endif
    __syncthreads();
    v16bf b0 = frag_b(sY, 32, 1, lane);
    v16bf b1 = frag_b(sY + 16, 32, 1, lane);
#pragma unroll
    for (int mt = 0; mt < 4; ++mt){
      int mrow = (wave * 4 + mt) * 16;
      v16bf a = frag_a(w3bf + mrow * 128 + kc * 32, 128, lane);
      acc[mt * 2]     = wmma_bf16(a, b0, acc[mt * 2]);
      acc[mt * 2 + 1] = wmma_bf16(a, b1, acc[mt * 2 + 1]);
    }
  }
  int half = lane >> 4;
#pragma unroll
  for (int mt = 0; mt < 4; ++mt){
#pragma unroll
    for (int nt = 0; nt < 2; ++nt){
      int nn = nt * 16 + (lane & 15), n = n0 + nn;
#pragma unroll
      for (int r = 0; r < 8; ++r){
        int gm = (wave * 4 + mt) * 16 + r + 8 * half;
        size_t idx = ((size_t)(b * COUT + gm)) * NPOS + n;
        float f = acc[mt * 2 + nt][r] * sS3[gm] + sB3[gm] + xin[idx];
        f = f > 0.0f ? f : 0.0f;
        out[idx] = f;
      }
    }
  }
}

// =====================================================================
extern "C" void kernel_launch(void* const* d_in, const int* in_sizes, int n_in,
                              void* d_out, int out_size, void* d_ws, size_t ws_size,
                              hipStream_t stream)
{
  const float* x       = (const float*)d_in[0];
  const float* conv1_w = (const float*)d_in[1];
  const float* bn1_g = (const float*)d_in[2];  const float* bn1_b = (const float*)d_in[3];
  const float* bn1_m = (const float*)d_in[4];  const float* bn1_v = (const float*)d_in[5];
  const float* q_w   = (const float*)d_in[6];
  const float* bnq_g = (const float*)d_in[7];  const float* bnq_b = (const float*)d_in[8];
  const float* bnq_m = (const float*)d_in[9];  const float* bnq_v = (const float*)d_in[10];
  const float* k_w   = (const float*)d_in[11];
  const float* v_w   = (const float*)d_in[12];
  const float* bnv_g = (const float*)d_in[13]; const float* bnv_b = (const float*)d_in[14];
  const float* bnv_m = (const float*)d_in[15]; const float* bnv_v = (const float*)d_in[16];
  const float* emb   = (const float*)d_in[17];
  const float* bn2_g = (const float*)d_in[18]; const float* bn2_b = (const float*)d_in[19];
  const float* bn2_m = (const float*)d_in[20]; const float* bn2_v = (const float*)d_in[21];
  const float* conv3_w = (const float*)d_in[22];
  const float* bn3_g = (const float*)d_in[23]; const float* bn3_b = (const float*)d_in[24];
  const float* bn3_m = (const float*)d_in[25]; const float* bn3_v = (const float*)d_in[26];

  char* ws = (char*)d_ws;
  unsigned short* w1bf   = (unsigned short*)(ws + OFF_W1);
  unsigned short* wqkvbf = (unsigned short*)(ws + OFF_WQKV);
  unsigned short* w3bf   = (unsigned short*)(ws + OFF_W3);
  unsigned short* embbf  = (unsigned short*)(ws + OFF_EMB);
  unsigned short* qws    = (unsigned short*)(ws + OFF_Q);
  unsigned short* kws    = (unsigned short*)(ws + OFF_K);
  float*          vfws   = (float*)         (ws + OFF_VF);
  float*          clws   = (float*)         (ws + OFF_CL);
  unsigned short* yws    = (unsigned short*)(ws + OFF_Y);
  float* outp = (float*)d_out;

  k_prep<<<584, 256, 0, stream>>>(conv1_w, q_w, k_w, v_w, conv3_w, emb,
                                  w1bf, wqkvbf, w3bf, embbf);
  k_fused_qkv<<<NB * (NPOS / 32), 256, 0, stream>>>(
      x, w1bf, wqkvbf,
      bn1_g, bn1_b, bn1_m, bn1_v,
      bnq_g, bnq_b, bnq_m, bnq_v,
      bnv_g, bnv_b, bnv_m, bnv_v,
      qws, kws, vfws);
  k_content_lambda<<<NB, 256, 0, stream>>>(kws, vfws, clws);
  k_lambda_apply<<<NB * 16, 256, 0, stream>>>(qws, vfws, embbf, clws,
                                              bn2_g, bn2_b, bn2_m, bn2_v, yws);
  k_conv3<<<NB * (NPOS / 32), 256, 0, stream>>>(yws, w3bf, x,
                                                bn3_g, bn3_b, bn3_m, bn3_v, outp);
}
